// GroupedQueryAttention_52183852647063
// MI455X (gfx1250) — matrix operand order repaired and checked by reading, not compiled, not run on hardware
//
#include <hip/hip_runtime.h>
#include <hip/hip_bf16.h>
#include <math.h>

// Problem constants (fixed by the reference)
#define BB 2
#define TT 2048
#define DD 1024
#define HH 16
#define HDD 64
#define NTOK 4096   // B*T

typedef __attribute__((ext_vector_type(16))) __bf16 v16bf;
typedef __attribute__((ext_vector_type(8)))  __bf16 v8bf;
typedef __attribute__((ext_vector_type(8)))  float  v8f;

static __device__ __forceinline__ __bf16 f2bf(float f) {
    union { float f; unsigned u; } v; v.f = f;
    unsigned r = v.u + 0x7FFFu + ((v.u >> 16) & 1u);   // round-to-nearest-even
    unsigned short h = (unsigned short)(r >> 16);
    __bf16 out;
    __builtin_memcpy(&out, &h, 2);
    return out;
}

static __device__ __forceinline__ v16bf cat16(v8bf lo, v8bf hi) {
    return __builtin_shufflevector(lo, hi, 0,1,2,3,4,5,6,7,8,9,10,11,12,13,14,15);
}

static __device__ __forceinline__ v8f wmma_bf16(v16bf a, v16bf b, v8f c) {
    // D(16x16 f32) = A(16x32 bf16) x B(32x16 bf16) + C
    return __builtin_amdgcn_wmma_f32_16x16x32_bf16(
        false, a, false, b, (short)0, c, false, false);
}

// A-frag: per lane K = {0..7, 16..23} relative to p
static __device__ __forceinline__ v16bf ldA(const __bf16* p) {
    return cat16(*(const v8bf*)p, *(const v8bf*)(p + 16));
}
// B-frag: per lane K = {0..7, 16..23} relative to p
static __device__ __forceinline__ v16bf ldB(const __bf16* p) {
    return cat16(*(const v8bf*)p, *(const v8bf*)(p + 16));
}

// gfx1250 async global->LDS copy (ASYNCcnt-tracked DMA, bypasses VGPRs)
static __device__ __forceinline__ unsigned lds_off_of(const void* p) {
    return (unsigned)(size_t)p;   // low 32 bits of generic LDS address = LDS offset
}
static __device__ __forceinline__ void async_ld_b128(unsigned lds_off, const void* g) {
    asm volatile("global_load_async_to_lds_b128 %0, %1, off"
                 :: "v"(lds_off), "v"((unsigned long long)(size_t)g)
                 : "memory");
}

// ---------------------------------------------------------------------------
// Pass 0a: f32 -> bf16 elementwise (activations)
__global__ void cvt_x_k(const float* __restrict__ x, __bf16* __restrict__ xb, int n) {
    int i = blockIdx.x * blockDim.x + threadIdx.x;
    int stride = gridDim.x * blockDim.x;
    for (; i < n; i += stride) xb[i] = f2bf(x[i]);
}

// Pass 0b: f32 W[k][n] -> bf16 W^T[n][k]  (1024x1024)
__global__ void cvt_wt_k(const float* __restrict__ W, __bf16* __restrict__ Wt) {
    int i = blockIdx.x * blockDim.x + threadIdx.x;      // 1024*1024 threads
    int nn = i >> 10, kk = i & 1023;
    Wt[i] = f2bf(W[kk * DD + nn]);
}

// ---------------------------------------------------------------------------
// Pass 1: fused QKV projection + bias + partial RoPE.
// 1 wave computes a 16(row) x 64(col) tile; K-loop is ping-pong pipelined
// (two fragment sets alternate -> no register-rotation moves).
// q,k stored bf16 [b*H+h][t][d]; v stored bf16 transposed [b*H+h][d][t].
__global__ __launch_bounds__(32) void qkv_proj_k(
    const __bf16* __restrict__ xb,
    const __bf16* __restrict__ Wqt, const __bf16* __restrict__ Wkt,
    const __bf16* __restrict__ Wvt,
    const float* __restrict__ bq, const float* __restrict__ bk,
    const float* __restrict__ bv,
    __bf16* __restrict__ qw, __bf16* __restrict__ kw, __bf16* __restrict__ vw)
{
    const int lane = threadIdx.x & 31;
    const int n  = lane & 15;
    const int hi = lane >> 4;
    const int gr0 = blockIdx.x * 16;      // token-row base
    const int c0  = blockIdx.y * 64;      // global col base over [Wq|Wk|Wv]
    const int mat = c0 >> 10;             // 0=q 1=k 2=v
    const int f0  = c0 & (DD - 1);        // col base within matrix

    const __bf16* Wt  = (mat == 0) ? Wqt : (mat == 1) ? Wkt : Wvt;
    const float* bias = (mat == 0) ? bq  : (mat == 1) ? bk  : bv;

    const __bf16* aP = xb + (size_t)(gr0 + n) * DD + hi * 8;
    const __bf16* bP[4];
#pragma unroll
    for (int nt = 0; nt < 4; nt++)
        bP[nt] = Wt + (size_t)(f0 + nt * 16 + n) * DD + hi * 8;

    v8f zero = {};
    v8f acc[4] = {zero, zero, zero, zero};

    // set 0 primed at k=0; sets ping-pong every 32 K
    v16bf a0 = ldA(aP);
    v16bf b0[4];
#pragma unroll
    for (int nt = 0; nt < 4; nt++) b0[nt] = ldB(bP[nt]);

    for (int k0 = 0; k0 < DD; k0 += 64) {
        __builtin_prefetch(aP + ((k0 + 256) & (DD - 1)), 0, 3);
        // half-iter A: load set1 @ k0+32, compute set0
        v16bf a1 = ldA(aP + k0 + 32);
        v16bf b1[4];
#pragma unroll
        for (int nt = 0; nt < 4; nt++) b1[nt] = ldB(bP[nt] + k0 + 32);
#pragma unroll
        for (int nt = 0; nt < 4; nt++) acc[nt] = wmma_bf16(a0, b0[nt], acc[nt]);
        // half-iter B: load set0 @ k0+64 (wraps on last iter), compute set1
        const int kn = (k0 + 64) & (DD - 1);
        a0 = ldA(aP + kn);
#pragma unroll
        for (int nt = 0; nt < 4; nt++) b0[nt] = ldB(bP[nt] + kn);
#pragma unroll
        for (int nt = 0; nt < 4; nt++) acc[nt] = wmma_bf16(a1, b1[nt], acc[nt]);
    }

    // Epilogue: bias + RoPE (first 32 dims of each head, q/k only), write bf16
    const int b  = gr0 / TT;
    const int h  = f0 >> 6;
    const int bh = b * HH + h;
#pragma unroll
    for (int nt = 0; nt < 4; nt++) {
        const int d = nt * 16 + n;                 // dim within head (0..63)
        const float bval = bias[f0 + d];
        const bool dorope = (mat < 2) && (d < 32);
        float invf = 0.f;
        if (dorope) invf = __powf(10000.0f, -(float)(d & ~1) * (1.0f / 32.0f));
        const bool even = ((d & 1) == 0);
#pragma unroll
        for (int r = 0; r < 8; r++) {
            const int m = r + 8 * hi;              // row within tile
            const int t = (gr0 + m) & (TT - 1);    // position in sequence
            float val = acc[nt][r] + bval;
            float partner = __shfl_xor(val, 1);    // pair element (d^1), same row
            float res = val;
            if (dorope) {
                float s, c;
                __sincosf((float)t * invf, &s, &c);
                res = even ? (val * c - partner * s) : (partner * s + val * c);
            }
            __bf16 ob = f2bf(res);
            if (mat == 2) {
                vw[((size_t)bh * HDD + d) * TT + t] = ob;          // transposed
            } else {
                __bf16* dst = (mat == 0) ? qw : kw;
                dst[((size_t)bh * TT + t) * HDD + d] = ob;
            }
        }
    }
}

// ---------------------------------------------------------------------------
// Pass 2: causal flash attention with async-DMA K/V staging.
// Block = 4 waves sharing one (b,h); wave w owns q-tile qt = by*4+w.
// Per 32-key block: K tile (32x64) and V tile (64x32, d-major) are DMA'd
// into LDS (double-buffered) by the whole block, then consumed via WMMA.
__global__ __launch_bounds__(128) void attn_k(
    const __bf16* __restrict__ qw, const __bf16* __restrict__ kw,
    const __bf16* __restrict__ vw, __bf16* __restrict__ ow)
{
    __shared__ __bf16 kbuf[2][32 * HDD];    // [key][d]
    __shared__ __bf16 vbuf[2][HDD * 32];    // [d][t]
    __shared__ __bf16 pbuf[4][16 * 32];     // per-wave P staging

    const int tid  = threadIdx.x;
    const int wave = tid >> 5;
    const int lane = tid & 31;
    const int n  = lane & 15;
    const int hi = lane >> 4;
    const int bh = blockIdx.x;
    const int qt = blockIdx.y * 4 + wave;

    const __bf16* qbase = qw + (size_t)bh * TT * HDD;
    const __bf16* kbase = kw + (size_t)bh * TT * HDD;
    const __bf16* vbase = vw + (size_t)bh * HDD * TT;
    __bf16* pl = pbuf[wave];

    // Q frags (reused for every key block): two K-steps of 32 over d=64
    v16bf qa[2];
#pragma unroll
    for (int s = 0; s < 2; s++) {
        const __bf16* qp = qbase + (size_t)(qt * 16 + n) * HDD + s * 32 + hi * 8;
        qa[s] = cat16(*(const v8bf*)qp, *(const v8bf*)(qp + 16));
    }

    v8f zero = {};
    v8f o[4] = {zero, zero, zero, zero};    // O: 16 x 64 f32
    float mi[8], li[8];
#pragma unroll
    for (int r = 0; r < 8; r++) { mi[r] = -INFINITY; li[r] = 0.0f; }

    const int jmax  = (qt * 16 + 15) >> 5;                    // own diagonal
    const int jmaxb = ((blockIdx.y * 4 + 3) * 16 + 15) >> 5;  // block-uniform bound

    // Stage key block jb into LDS buffer `buf` (4 async b128 per thread)
    auto stage = [&](int jb, int buf) {
        // K tile: 4 KB contiguous ([t][d] layout, 32 consecutive keys)
        {
            const char* g = (const char*)(kbase + (size_t)jb * 32 * HDD) + tid * 32;
            unsigned    l = lds_off_of(&kbuf[buf][0]) + (unsigned)(tid * 32);
            async_ld_b128(l,      g);
            async_ld_b128(l + 16, g + 16);
        }
        // V tile: 64 rows (d) x 64 B; each thread moves half a row
        {
            const int d = tid >> 1, half = tid & 1;
            const char* g = (const char*)(vbase + (size_t)d * TT + jb * 32 + half * 16);
            unsigned    l = lds_off_of(&vbuf[buf][0]) + (unsigned)(d * 64 + half * 32);
            async_ld_b128(l,      g);
            async_ld_b128(l + 16, g + 16);
        }
    };

    stage(0, 0);
    for (int jb = 0; jb <= jmaxb; jb++) {
        const int cur = jb & 1;
        if (jb < jmaxb) {
            stage(jb + 1, cur ^ 1);                          // prefetch next tile
            asm volatile("s_wait_asynccnt 0x4" ::: "memory"); // cur tile landed
        } else {
            asm volatile("s_wait_asynccnt 0x0" ::: "memory");
        }
        __syncthreads();

        if (jb <= jmax) {   // wave-uniform: EXEC stays all-1s for WMMA
            const __bf16* kb = &kbuf[cur][0];
            const __bf16* vb = &vbuf[cur][0];

            // S = Q @ K^T for 32 keys -> two 16x16 tiles (K frags from LDS)
            v8f s0 = zero, s1 = zero;
#pragma unroll
            for (int s = 0; s < 2; s++) {
                const __bf16* kp0 = kb + (n)      * HDD + s * 32 + hi * 8;
                const __bf16* kp1 = kb + (16 + n) * HDD + s * 32 + hi * 8;
                s0 = wmma_bf16(qa[s], ldB(kp0), s0);
                s1 = wmma_bf16(qa[s], ldB(kp1), s1);
            }
            // scale + causal mask + online softmax
            float p0[8], p1[8];
#pragma unroll
            for (int r = 0; r < 8; r++) {
                const int qidx = qt * 16 + r + 8 * hi;
                const int kidx = jb * 32 + n;
                p0[r] = (kidx      <= qidx) ? s0[r] * 0.125f : -INFINITY;
                p1[r] = (kidx + 16 <= qidx) ? s1[r] * 0.125f : -INFINITY;
            }
#pragma unroll
            for (int r = 0; r < 8; r++) {
                float rm = fmaxf(p0[r], p1[r]);
                rm = fmaxf(rm, __shfl_xor(rm, 1));
                rm = fmaxf(rm, __shfl_xor(rm, 2));
                rm = fmaxf(rm, __shfl_xor(rm, 4));
                rm = fmaxf(rm, __shfl_xor(rm, 8));
                float mnew  = fmaxf(mi[r], rm);
                float alpha = __expf(mi[r] - mnew);
                float e0 = __expf(p0[r] - mnew);
                float e1 = __expf(p1[r] - mnew);
                float rs = e0 + e1;
                rs += __shfl_xor(rs, 1);
                rs += __shfl_xor(rs, 2);
                rs += __shfl_xor(rs, 4);
                rs += __shfl_xor(rs, 8);
                li[r] = li[r] * alpha + rs;
                mi[r] = mnew;
                o[0][r] *= alpha; o[1][r] *= alpha; o[2][r] *= alpha; o[3][r] *= alpha;
                p0[r] = e0; p1[r] = e1;
            }
            // P (C-layout) -> LDS -> A-layout frag
#pragma unroll
            for (int r = 0; r < 8; r++) {
                const int m = r + 8 * hi;
                pl[m * 32 + n]      = f2bf(p0[r]);
                pl[m * 32 + 16 + n] = f2bf(p1[r]);
            }
            asm volatile("s_wait_dscnt 0x0" ::: "memory");
            v16bf pa = cat16(*(const v8bf*)(pl + n * 32 + hi * 8),
                             *(const v8bf*)(pl + n * 32 + hi * 8 + 16));
            // O += P @ V  (V frags from LDS, d-major => contiguous over t)
#pragma unroll
            for (int nt = 0; nt < 4; nt++) {
                const __bf16* vp = vb + (nt * 16 + n) * 32 + hi * 8;
                o[nt] = wmma_bf16(pa, ldB(vp), o[nt]);
            }
        }
        __syncthreads();    // all reads of `cur` done before it is re-staged
    }

    // Epilogue: O / l, store bf16 [b][t][h*64+d] for the output GEMM
    const int b = bh >> 4, h = bh & 15;
#pragma unroll
    for (int nt = 0; nt < 4; nt++) {
#pragma unroll
        for (int r = 0; r < 8; r++) {
            const int m = r + 8 * hi;
            const int t = qt * 16 + m;
            ow[((size_t)(b * TT + t)) * DD + h * HDD + nt * 16 + n] =
                f2bf(o[nt][r] / li[r]);
        }
    }
}

// ---------------------------------------------------------------------------
// Pass 3: output projection attn(4096x1024 bf16) @ Wo + bo -> f32 d_out
// Same ping-pong pipelined GEMM structure as Pass 1.
__global__ __launch_bounds__(32) void out_proj_k(
    const __bf16* __restrict__ ab, const __bf16* __restrict__ Wot,
    const float* __restrict__ bo, float* __restrict__ out)
{
    const int lane = threadIdx.x & 31;
    const int n  = lane & 15;
    const int hi = lane >> 4;
    const int gr0 = blockIdx.x * 16;
    const int c0  = blockIdx.y * 64;

    const __bf16* aP = ab + (size_t)(gr0 + n) * DD + hi * 8;
    const __bf16* bP[4];
#pragma unroll
    for (int nt = 0; nt < 4; nt++)
        bP[nt] = Wot + (size_t)(c0 + nt * 16 + n) * DD + hi * 8;

    v8f zero = {};
    v8f acc[4] = {zero, zero, zero, zero};

    v16bf a0 = ldA(aP);
    v16bf b0[4];
#pragma unroll
    for (int nt = 0; nt < 4; nt++) b0[nt] = ldB(bP[nt]);

    for (int k0 = 0; k0 < DD; k0 += 64) {
        __builtin_prefetch(aP + ((k0 + 256) & (DD - 1)), 0, 3);
        v16bf a1 = ldA(aP + k0 + 32);
        v16bf b1[4];
#pragma unroll
        for (int nt = 0; nt < 4; nt++) b1[nt] = ldB(bP[nt] + k0 + 32);
#pragma unroll
        for (int nt = 0; nt < 4; nt++) acc[nt] = wmma_bf16(a0, b0[nt], acc[nt]);
        const int kn = (k0 + 64) & (DD - 1);
        a0 = ldA(aP + kn);
#pragma unroll
        for (int nt = 0; nt < 4; nt++) b0[nt] = ldB(bP[nt] + kn);
#pragma unroll
        for (int nt = 0; nt < 4; nt++) acc[nt] = wmma_bf16(a1, b1[nt], acc[nt]);
    }
#pragma unroll
    for (int nt = 0; nt < 4; nt++) {
        const float bval = bo[c0 + nt * 16 + n];
#pragma unroll
        for (int r = 0; r < 8; r++) {
            const int m = r + 8 * hi;
            out[(size_t)(gr0 + m) * DD + c0 + nt * 16 + n] = acc[nt][r] + bval;
        }
    }
}

// ---------------------------------------------------------------------------
extern "C" void kernel_launch(void* const* d_in, const int* in_sizes, int n_in,
                              void* d_out, int out_size, void* d_ws, size_t ws_size,
                              hipStream_t stream) {
    const float* x  = (const float*)d_in[0];
    const float* Wq = (const float*)d_in[1];
    const float* bq = (const float*)d_in[2];
    const float* Wk = (const float*)d_in[3];
    const float* bk = (const float*)d_in[4];
    const float* Wv = (const float*)d_in[5];
    const float* bv = (const float*)d_in[6];
    const float* Wo = (const float*)d_in[7];
    const float* bo = (const float*)d_in[8];

    // Workspace layout (48 MB total)
    char* ws = (char*)d_ws;
    const size_t MB = 1u << 20;
    __bf16* xb  = (__bf16*)(ws);              //  8 MB: x bf16
    __bf16* Wqt = (__bf16*)(ws +  8 * MB);    //  2 MB each: W^T bf16
    __bf16* Wkt = (__bf16*)(ws + 10 * MB);
    __bf16* Wvt = (__bf16*)(ws + 12 * MB);
    __bf16* Wot = (__bf16*)(ws + 14 * MB);
    __bf16* qw  = (__bf16*)(ws + 16 * MB);    //  8 MB: q  [bh][t][d]
    __bf16* kw  = (__bf16*)(ws + 24 * MB);    //  8 MB: k  [bh][t][d]
    __bf16* vw  = (__bf16*)(ws + 32 * MB);    //  8 MB: v  [bh][d][t]
    __bf16* aw  = (__bf16*)(ws + 40 * MB);    //  8 MB: attn out [b][t][h*64+d]

    // Pass 0: conversions / transposes
    cvt_x_k<<<16384, 256, 0, stream>>>(x, xb, NTOK * DD);
    cvt_wt_k<<<4096, 256, 0, stream>>>(Wq, Wqt);
    cvt_wt_k<<<4096, 256, 0, stream>>>(Wk, Wkt);
    cvt_wt_k<<<4096, 256, 0, stream>>>(Wv, Wvt);
    cvt_wt_k<<<4096, 256, 0, stream>>>(Wo, Wot);

    // Pass 1: QKV projection + RoPE
    qkv_proj_k<<<dim3(NTOK / 16, 3 * DD / 64), 32, 0, stream>>>(
        xb, Wqt, Wkt, Wvt, bq, bk, bv, qw, kw, vw);

    // Pass 2: causal flash attention (async K/V staging)
    attn_k<<<dim3(BB * HH, TT / 16 / 4), 128, 0, stream>>>(qw, kw, vw, aw);

    // Pass 3: output projection
    out_proj_k<<<dim3(NTOK / 16, DD / 64), 32, 0, stream>>>(aw, Wot, bo, (float*)d_out);

    (void)in_sizes; (void)n_in; (void)out_size; (void)ws_size;
}
